// GroupedQueryAttention_48902497632837
// MI455X (gfx1250) — matrix operand order repaired and checked by reading, not compiled, not run on hardware
//
#include <hip/hip_runtime.h>

typedef unsigned short u16;
typedef __attribute__((ext_vector_type(16))) __bf16 bfx16;
typedef __attribute__((ext_vector_type(8)))  float  fx8;

typedef int v4i __attribute__((vector_size(16)));
typedef v4i __attribute__((address_space(1))) *gv4i_p;
typedef v4i __attribute__((address_space(3))) *lv4i_p;

struct Pack32 { uint4 a, b; };

__device__ __forceinline__ u16 f2bf(float f) {
    unsigned u = __float_as_uint(f);
    u += 0x7fffu + ((u >> 16) & 1u);            // round-to-nearest-even
    return (u16)(u >> 16);
}

__device__ __forceinline__ bfx16 ld_frag2(const u16* p0, const u16* p1) {
    Pack32 t;
    t.a = *reinterpret_cast<const uint4*>(p0);
    t.b = *reinterpret_cast<const uint4*>(p1);
    return __builtin_bit_cast(bfx16, t);
}
__device__ __forceinline__ bfx16 ld_frag32(const u16* p) { return ld_frag2(p, p + 8); }

__device__ __forceinline__ fx8 wmma_bf16(bfx16 a, bfx16 b, fx8 c) {
    return __builtin_amdgcn_wmma_f32_16x16x32_bf16(false, a, false, b, (short)0, c, false, false);
}

// --- CDNA5 async global->LDS (ASYNCcnt) with safe fallback -----------------
__device__ __forceinline__ void async_ld_b128(const void* g, void* l) {
#if __has_builtin(__builtin_amdgcn_global_load_async_to_lds_b128)
    __builtin_amdgcn_global_load_async_to_lds_b128(
        (gv4i_p)(unsigned long long)g,
        (lv4i_p)(unsigned)(unsigned long long)l, 0, 0);
#else
    *reinterpret_cast<uint4*>(l) = *reinterpret_cast<const uint4*>(g);
#endif
}

__device__ __forceinline__ void wait_async0() {
#if __has_builtin(__builtin_amdgcn_s_wait_asynccnt)
    __builtin_amdgcn_s_wait_asynccnt(0);
#else
    asm volatile("s_wait_asynccnt 0" ::: "memory");
#endif
}

// ---------------------------------------------------------------- casts ----
__global__ void cast_f32_bf16(const float* __restrict__ src, u16* __restrict__ dst, int n) {
    int i = blockIdx.x * blockDim.x + threadIdx.x;
    if (i < n) dst[i] = f2bf(src[i]);
}

// ----------------------------------------------------------------- GEMM ----
// C[M,N] = A[M,K] * B[K,N], A/B bf16 row-major; OUT_BF16 selects output type.
// Block: 256 thr (8 waves, 4x2). Tile 128x128, BK=32. Wave: 2x4 WMMA tiles.
// Double-buffered LDS; A staged via async DMA, B transposed via VALU scatter.
#define GBM 128
#define GBN 128
#define GBK 32

template <bool OUT_BF16>
__global__ __launch_bounds__(256)
void gemm_wmma(const u16* __restrict__ A, const u16* __restrict__ B,
               void* __restrict__ Cv, int M, int N, int K) {
    __shared__ u16 As[2][GBM][GBK + 8];   // 2 x 128 x 40 (row = 80 B, 16B aligned)
    __shared__ u16 Bt[2][GBN][GBK + 8];   // B transposed: [n][k]

    const int tid  = threadIdx.x;
    const int lane = tid & 31, wave = tid >> 5;
    const int wm = wave >> 1, wn = wave & 1;         // 4 x 2 wave grid
    const int l  = lane & 15, hi = lane >> 4;
    const int tileM = blockIdx.y * GBM, tileN = blockIdx.x * GBN;

    auto load_tile = [&](int k0, int buf) {
        // A tile 128x32: 512 x 16B chunks, 2/thread, async DMA to LDS
#pragma unroll
        for (int c = tid; c < (GBM * GBK) / 8; c += 256) {
            int row = c >> 2, cg = c & 3;
            async_ld_b128(&A[(size_t)(tileM + row) * K + k0 + cg * 8],
                          &As[buf][row][cg * 8]);
        }
        // B tile 32x128 -> transposed scatter: 512 chunks, 2/thread
#pragma unroll
        for (int c = tid; c < (GBK * GBN) / 8; c += 256) {
            int kk = c >> 4, ng = c & 15;
            union { uint4 v; u16 s[8]; } u;
            u.v = *reinterpret_cast<const uint4*>(&B[(size_t)(k0 + kk) * N + tileN + ng * 8]);
#pragma unroll
            for (int j = 0; j < 8; j++) Bt[buf][ng * 8 + j][kk] = u.s[j];
        }
    };

    fx8 acc[2][4];
#pragma unroll
    for (int mi = 0; mi < 2; mi++)
#pragma unroll
        for (int ni = 0; ni < 4; ni++)
            acc[mi][ni] = (fx8){0.f, 0.f, 0.f, 0.f, 0.f, 0.f, 0.f, 0.f};

    load_tile(0, 0);
    wait_async0();
    __syncthreads();

    const int nk = K / GBK;
    int buf = 0;
    for (int kt = 0; kt < nk; kt++) {
        if (kt + 1 < nk) load_tile((kt + 1) * GBK, buf ^ 1);  // prefetch next tile

        bfx16 af[2], bf[4];
#pragma unroll
        for (int mi = 0; mi < 2; mi++) {
            const u16* r = &As[buf][wm * 32 + mi * 16 + l][0];
            af[mi] = ld_frag2(r + hi * 8, r + 16 + hi * 8);
        }
#pragma unroll
        for (int ni = 0; ni < 4; ni++) {
            const u16* r = &Bt[buf][wn * 64 + ni * 16 + l][0];
            bf[ni] = ld_frag2(r + hi * 8, r + 16 + hi * 8);
        }
#pragma unroll
        for (int mi = 0; mi < 2; mi++)
#pragma unroll
            for (int ni = 0; ni < 4; ni++)
                acc[mi][ni] = wmma_bf16(af[mi], bf[ni], acc[mi][ni]);

        wait_async0();      // next tile's DMA complete
        __syncthreads();    // all waves done reading buf / writing buf^1
        buf ^= 1;
    }

#pragma unroll
    for (int mi = 0; mi < 2; mi++)
#pragma unroll
        for (int ni = 0; ni < 4; ni++)
#pragma unroll
            for (int i = 0; i < 8; i++) {
                int row = tileM + wm * 32 + mi * 16 + hi * 8 + i;
                int col = tileN + wn * 64 + ni * 16 + l;
                float v = acc[mi][ni][i];
                if (OUT_BF16) ((u16*)Cv)[(size_t)row * N + col] = f2bf(v);
                else          ((float*)Cv)[(size_t)row * N + col] = v;
            }
}

// ----------------------------------------------------------------- RoPE ----
__global__ void rope_cast(const float* __restrict__ X, u16* __restrict__ Y,
                          int T, int H, int total) {
    int idx = blockIdx.x * blockDim.x + threadIdx.x;
    if (idx >= total) return;
    int i  = idx & 63;
    int h  = (idx >> 6) % H;
    int bt = idx / (64 * H);
    int t  = bt % T;
    size_t base = (size_t)bt * (H * 128) + h * 128 + i;
    float a = X[base], b = X[base + 64];
    float ang = (float)t * __expf(-(float)i * 0.14391156f);  // ln(1e4)/64
    float sn, cs;
    __sincosf(ang, &sn, &cs);
    Y[base]      = f2bf(a * cs - b * sn);
    Y[base + 64] = f2bf(b * cs + a * sn);
}

// ------------------------------------------------------- flash attention ----
#define FA_SCALE 0.08838834764831845f   // 128^-0.5

__global__ __launch_bounds__(256)
void fa_kernel(const u16* __restrict__ Qb, const u16* __restrict__ Kb,
               const u16* __restrict__ Vb, u16* __restrict__ AO, int T) {
    __shared__ u16 Ks[64][128 + 8];     // [key][dim]
    __shared__ u16 Vt[128][64 + 8];     // [dim][key]
    __shared__ u16 Ps[8][16][72];       // per-wave P tile [row][key]

    const int b = blockIdx.z, h = blockIdx.y, kvh = h >> 2;
    const int q0 = blockIdx.x * 128;
    const int tid = threadIdx.x, lane = tid & 31, wave = tid >> 5;
    const int l = lane & 15, hi = lane >> 4;
    const int DQ = 2048, DKV = 512;

    const u16* Qp = Qb + (size_t)b * T * DQ  + (size_t)h   * 128;
    const u16* Kp = Kb + (size_t)b * T * DKV + (size_t)kvh * 128;
    const u16* Vp = Vb + (size_t)b * T * DKV + (size_t)kvh * 128;

    bfx16 qf[4];
    {
        const u16* qr = Qp + (size_t)(q0 + wave * 16 + l) * DQ;
#pragma unroll
        for (int kc = 0; kc < 4; kc++)
            qf[kc] = ld_frag2(qr + kc * 32 + hi * 8, qr + kc * 32 + 16 + hi * 8);
    }

    fx8 o[8];
#pragma unroll
    for (int nj = 0; nj < 8; nj++) o[nj] = (fx8){0.f, 0.f, 0.f, 0.f, 0.f, 0.f, 0.f, 0.f};
    float run_m[8], run_l[8];
#pragma unroll
    for (int i = 0; i < 8; i++) { run_m[i] = -__builtin_inff(); run_l[i] = 0.f; }

    const int qbase = q0 + wave * 16 + hi * 8;
    const int ntiles = (q0 + 128) / 64;

    for (int tt = 0; tt < ntiles; tt++) {
        const int kv0 = tt * 64;
        __syncthreads();
        // K tile 64x128: async DMA, 4 chunks/thread
#pragma unroll
        for (int c = tid; c < 1024; c += 256) {
            int key = c >> 4, dg = c & 15;
            async_ld_b128(&Kp[(size_t)(kv0 + key) * DKV + dg * 8], &Ks[key][dg * 8]);
        }
        // V tile transposed (VALU scatter)
#pragma unroll
        for (int c = tid; c < 1024; c += 256) {
            int key = c >> 4, dg = c & 15;
            union { uint4 v; u16 s[8]; } u;
            u.v = *reinterpret_cast<const uint4*>(&Vp[(size_t)(kv0 + key) * DKV + dg * 8]);
#pragma unroll
            for (int j = 0; j < 8; j++) Vt[dg * 8 + j][key] = u.s[j];
        }
        wait_async0();
        __syncthreads();

        // S = Q K^T : 4 n-tiles x 4 k-chunks
        fx8 s[4];
#pragma unroll
        for (int ni = 0; ni < 4; ni++) {
            s[ni] = (fx8){0.f, 0.f, 0.f, 0.f, 0.f, 0.f, 0.f, 0.f};
#pragma unroll
            for (int kc = 0; kc < 4; kc++) {
                bfx16 kf = ld_frag2(&Ks[ni * 16 + l][kc * 32 + hi * 8], &Ks[ni * 16 + l][kc * 32 + 16 + hi * 8]);
                s[ni] = wmma_bf16(qf[kc], kf, s[ni]);
            }
        }

        // online softmax (rows hi*8+i; cols ni*16+l)
#pragma unroll
        for (int i = 0; i < 8; i++) {
            float mv = -__builtin_inff();
#pragma unroll
            for (int ni = 0; ni < 4; ni++) {
                float v = s[ni][i] * FA_SCALE;
                int key = kv0 + ni * 16 + l;
                v = (key <= qbase + i) ? v : -__builtin_inff();
                s[ni][i] = v;
                mv = fmaxf(mv, v);
            }
#pragma unroll
            for (int m = 1; m < 16; m <<= 1) mv = fmaxf(mv, __shfl_xor(mv, m, 32));
            float newm  = fmaxf(run_m[i], mv);
            float alpha = __expf(run_m[i] - newm);
            float rsum  = 0.f;
#pragma unroll
            for (int ni = 0; ni < 4; ni++) {
                float p = __expf(s[ni][i] - newm);
                s[ni][i] = p;
                rsum += p;
            }
#pragma unroll
            for (int m = 1; m < 16; m <<= 1) rsum += __shfl_xor(rsum, m, 32);
            run_l[i] = run_l[i] * alpha + rsum;
            run_m[i] = newm;
#pragma unroll
            for (int nj = 0; nj < 8; nj++) o[nj][i] *= alpha;
        }

        // stage P (C-layout -> LDS), reload as A-fragments
#pragma unroll
        for (int i = 0; i < 8; i++)
#pragma unroll
            for (int ni = 0; ni < 4; ni++)
                Ps[wave][hi * 8 + i][ni * 16 + l] = f2bf(s[ni][i]);
        __syncthreads();

        // O += P V : 2 k-chunks x 8 n-tiles
#pragma unroll
        for (int kc = 0; kc < 2; kc++) {
            const u16* pr = &Ps[wave][l][0];
            bfx16 pf = ld_frag2(pr + kc * 32 + hi * 8, pr + kc * 32 + 16 + hi * 8);
#pragma unroll
            for (int nj = 0; nj < 8; nj++) {
                bfx16 vf = ld_frag2(&Vt[nj * 16 + l][kc * 32 + hi * 8], &Vt[nj * 16 + l][kc * 32 + 16 + hi * 8]);
                o[nj] = wmma_bf16(pf, vf, o[nj]);
            }
        }
    }

    u16* AOp = AO + (size_t)b * T * DQ + (size_t)h * 128;
#pragma unroll
    for (int i = 0; i < 8; i++) {
        float inv = 1.0f / run_l[i];
        int row = q0 + wave * 16 + hi * 8 + i;
#pragma unroll
        for (int nj = 0; nj < 8; nj++)
            AOp[(size_t)row * DQ + nj * 16 + l] = f2bf(o[nj][i] * inv);
    }
}

// --------------------------------------------------------------- launch ----
extern "C" void kernel_launch(void* const* d_in, const int* in_sizes, int n_in,
                              void* d_out, int out_size, void* d_ws, size_t ws_size,
                              hipStream_t stream) {
    (void)in_sizes; (void)n_in; (void)out_size; (void)ws_size;
    const float* x  = (const float*)d_in[0];
    const float* Wq = (const float*)d_in[1];
    const float* Wk = (const float*)d_in[2];
    const float* Wv = (const float*)d_in[3];
    const float* Wo = (const float*)d_in[4];
    float* out = (float*)d_out;

    const int B = 2, T = 2048, D = 2048, BT = B * T;

    char* ws = (char*)d_ws;
    size_t off = 0;
    auto alloc = [&](size_t bytes) -> char* {
        off = (off + 255) & ~(size_t)255;
        char* p = ws + off;
        off += bytes;
        return p;
    };
    u16*   xb  = (u16*)alloc((size_t)BT * D * 2);
    u16*   Wqb = (u16*)alloc((size_t)D * 2048 * 2);
    u16*   Wkb = (u16*)alloc((size_t)D * 512 * 2);
    u16*   Wvb = (u16*)alloc((size_t)D * 512 * 2);
    u16*   Wob = (u16*)alloc((size_t)D * D * 2);
    float* Qf  = (float*)alloc((size_t)BT * 2048 * 4);
    float* Kf  = (float*)alloc((size_t)BT * 512 * 4);
    u16*   Qr  = (u16*)alloc((size_t)BT * 2048 * 2);
    u16*   Kr  = (u16*)alloc((size_t)BT * 512 * 2);
    u16*   Vr  = (u16*)alloc((size_t)BT * 512 * 2);
    u16*   AO  = (u16*)Qf;   // reuse: Qf dead after RoPE

    auto cgrid = [](int n) { return (n + 255) / 256; };
    cast_f32_bf16<<<cgrid(BT * D),   256, 0, stream>>>(x,  xb,  BT * D);
    cast_f32_bf16<<<cgrid(D * 2048), 256, 0, stream>>>(Wq, Wqb, D * 2048);
    cast_f32_bf16<<<cgrid(D * 512),  256, 0, stream>>>(Wk, Wkb, D * 512);
    cast_f32_bf16<<<cgrid(D * 512),  256, 0, stream>>>(Wv, Wvb, D * 512);
    cast_f32_bf16<<<cgrid(D * D),    256, 0, stream>>>(Wo, Wob, D * D);

    gemm_wmma<false><<<dim3(2048 / GBN, BT / GBM), 256, 0, stream>>>(xb, Wqb, Qf, BT, 2048, D);
    gemm_wmma<false><<<dim3(512  / GBN, BT / GBM), 256, 0, stream>>>(xb, Wkb, Kf, BT, 512,  D);
    gemm_wmma<true ><<<dim3(512  / GBN, BT / GBM), 256, 0, stream>>>(xb, Wvb, Vr, BT, 512,  D);

    int totQ = BT * 16 * 64, totK = BT * 4 * 64;
    rope_cast<<<cgrid(totQ), 256, 0, stream>>>(Qf, Qr, T, 16, totQ);
    rope_cast<<<cgrid(totK), 256, 0, stream>>>(Kf, Kr, T, 4,  totK);

    fa_kernel<<<dim3(T / 128, 16, B), 256, 0, stream>>>(Qr, Kr, Vr, AO, T);

    gemm_wmma<false><<<dim3(2048 / GBN, BT / GBM), 256, 0, stream>>>(AO, Wob, out, BT, 2048, D);
}
